// PairwiseRelaxedWordMoverSimilarity_30709016167128
// MI455X (gfx1250) — matrix operand order repaired and checked by reading, not compiled, not run on hardware
//
#include <hip/hip_runtime.h>
#include <hip/hip_bf16.h>
#include <math.h>
#include <stdint.h>

typedef __attribute__((ext_vector_type(16))) _Float16 v16h;
typedef __attribute__((ext_vector_type(8)))  _Float16 v8h;
typedef __attribute__((ext_vector_type(8)))  float    v8f;
typedef __attribute__((ext_vector_type(4)))  int      v4i;

#define AS1 __attribute__((address_space(1)))
#define AS3 __attribute__((address_space(3)))

static constexpr int HD = 768;          // hidden dim
static constexpr int SD = 128;          // sequence length
static constexpr int BD = 64;           // batch (both sides)
static constexpr int KT = 32;           // k-tile (one WMMA K)
static constexpr int NKT = HD / KT;     // 24 k-steps
static constexpr int LST = KT + 8;      // padded LDS row stride (f16) -> bank-conflict free

// CDNA5 async global->LDS path (ASYNCcnt), with portable fallback.
#if defined(__has_builtin)
# if __has_builtin(__builtin_amdgcn_global_load_async_to_lds_b128) && \
     __has_builtin(__builtin_amdgcn_s_wait_asynccnt)
#  define USE_ASYNC_LDS 1
# endif
#endif
#ifndef USE_ASYNC_LDS
# define USE_ASYNC_LDS 0
#endif

// ---------------------------------------------------------------------------
// Kernel 1: L2-normalize each token vector, emit f16 (one block per token).
// ---------------------------------------------------------------------------
__global__ __launch_bounds__(256) void normalize_f16_kernel(
    const float* __restrict__ x, _Float16* __restrict__ y)
{
    const int t = blockIdx.x;
    const float* row = x + (size_t)t * HD;
    _Float16* out   = y + (size_t)t * HD;

    float ss = 0.f;
    for (int i = threadIdx.x; i < HD; i += 256) {
        float v = row[i];
        ss += v * v;
    }
    #pragma unroll
    for (int d = 16; d >= 1; d >>= 1) ss += __shfl_xor(ss, d, 32);

    __shared__ float wsum[8];
    __shared__ float s_inv;
    const int wave = threadIdx.x >> 5;
    if ((threadIdx.x & 31) == 0) wsum[wave] = ss;
    __syncthreads();
    if (threadIdx.x == 0) {
        float tot = 0.f;
        #pragma unroll
        for (int w = 0; w < 8; ++w) tot += wsum[w];
        s_inv = rsqrtf(fmaxf(tot, 1e-16f));
    }
    __syncthreads();
    const float inv = s_inv;
    for (int i = threadIdx.x; i < HD; i += 256)
        out[i] = (_Float16)(row[i] * inv);
}

// ---------------------------------------------------------------------------
// Kernel 2: one block per (b1,b2). A/B k-chunks staged in LDS once per block
// (double-buffered, async global->LDS when available), 8 waves x 8 WMMA tiles,
// fused masked row/col max + masked means.
// ---------------------------------------------------------------------------
__global__ __launch_bounds__(256) void rwmd_pair_kernel(
    const _Float16* __restrict__ y1, const _Float16* __restrict__ y2,
    const int* __restrict__ mask1, const int* __restrict__ mask2,
    float* __restrict__ out)
{
    const int b1   = blockIdx.x;
    const int b2   = blockIdx.y;
    const int tid  = threadIdx.x;
    const int wave = tid >> 5;
    const int lane = tid & 31;
    const int half = lane >> 4;
    const int l16  = lane & 15;

    const int* __restrict__ m1row = mask1 + b1 * SD;
    const int* __restrict__ m2row = mask2 + b2 * SD;

    __shared__ _Float16 sA[2][SD * LST];   // 2 x 10 KB
    __shared__ _Float16 sB[2][SD * LST];   // 2 x 10 KB
    __shared__ float rowpart[16];
    __shared__ float colpart[8][SD];
    __shared__ float colval[SD];
    __shared__ int s_cnt1, s_cnt2;

    if (tid == 0) { s_cnt1 = 0; s_cnt2 = 0; }
    __syncthreads();
    if (tid < SD) {
        atomicAdd(&s_cnt1, m1row[tid]);
        atomicAdd(&s_cnt2, m2row[tid]);
    }

    const _Float16* __restrict__ Abase = y1 + (size_t)b1 * SD * HD;
    const _Float16* __restrict__ Bbase = y2 + (size_t)b2 * SD * HD;

    // Each thread moves two 16-byte chunks per matrix per k-step.
    const int c0 = tid, c1 = tid + 256;               // chunk ids, 512 per matrix
    const int row0 = c0 >> 2, off0 = (c0 & 3) * 8;    // f16 offsets
    const int row1 = c1 >> 2, off1 = (c1 & 3) * 8;

    auto stage = [&](int k0, int bb) {
#if USE_ASYNC_LDS
        __builtin_amdgcn_global_load_async_to_lds_b128(
            (AS1 v4i*)(uintptr_t)(Abase + (size_t)row0 * HD + k0 + off0),
            (AS3 v4i*)(uintptr_t)&sA[bb][row0 * LST + off0], 0, 0);
        __builtin_amdgcn_global_load_async_to_lds_b128(
            (AS1 v4i*)(uintptr_t)(Abase + (size_t)row1 * HD + k0 + off1),
            (AS3 v4i*)(uintptr_t)&sA[bb][row1 * LST + off1], 0, 0);
        __builtin_amdgcn_global_load_async_to_lds_b128(
            (AS1 v4i*)(uintptr_t)(Bbase + (size_t)row0 * HD + k0 + off0),
            (AS3 v4i*)(uintptr_t)&sB[bb][row0 * LST + off0], 0, 0);
        __builtin_amdgcn_global_load_async_to_lds_b128(
            (AS1 v4i*)(uintptr_t)(Bbase + (size_t)row1 * HD + k0 + off1),
            (AS3 v4i*)(uintptr_t)&sB[bb][row1 * LST + off1], 0, 0);
#else
        *(uint4*)&sA[bb][row0 * LST + off0] = *(const uint4*)(Abase + (size_t)row0 * HD + k0 + off0);
        *(uint4*)&sA[bb][row1 * LST + off1] = *(const uint4*)(Abase + (size_t)row1 * HD + k0 + off1);
        *(uint4*)&sB[bb][row0 * LST + off0] = *(const uint4*)(Bbase + (size_t)row0 * HD + k0 + off0);
        *(uint4*)&sB[bb][row1 * LST + off1] = *(const uint4*)(Bbase + (size_t)row1 * HD + k0 + off1);
#endif
    };

    v8f acc[8];
    #pragma unroll
    for (int j = 0; j < 8; ++j) acc[j] = (v8f){};

    // LDS fragment offsets (f16 units).
    // A (16x32): lanes 0-15 K{0..7,16..23}, lanes 16-31 K{8..15,24..31}.
    const int aoff = (16 * wave + l16) * LST + 8 * half;
    // B (32x16): lane = column N, lanes 0-15 K{0..7,16..23}, lanes 16-31 K{8..15,24..31}.
    const int boff = l16 * LST + 8 * half;

    stage(0, 0);

    for (int kt = 0; kt < NKT; ++kt) {
        const int bb = kt & 1;
        if (kt + 1 < NKT) {
            stage((kt + 1) * KT, bb ^ 1);
#if USE_ASYNC_LDS
            __builtin_amdgcn_s_wait_asynccnt(4);   // retire chunk kt's 4 transfers
#endif
        }
#if USE_ASYNC_LDS
        else {
            __builtin_amdgcn_s_wait_asynccnt(0);
        }
#endif
        __syncthreads();                           // chunk kt fully in LDS

        union { v16h v; v8h h[2]; } af;
        af.h[0] = *(const v8h*)&sA[bb][aoff];
        af.h[1] = *(const v8h*)&sA[bb][aoff + 16];
        #pragma unroll
        for (int j = 0; j < 8; ++j) {
            union { v16h v; v8h h[2]; } bf;
            bf.h[0] = *(const v8h*)&sB[bb][boff + j * 16 * LST];
            bf.h[1] = *(const v8h*)&sB[bb][boff + j * 16 * LST + 16];
            acc[j] = __builtin_amdgcn_wmma_f32_16x16x32_f16(
                false, af.v, false, bf.v, (short)0, acc[j], false, false);
        }
        __syncthreads();                           // all waves done before overwrite
    }

    // C layout: acc[j] component r <-> sim[m, n], m = 16*wave + r + 8*half,
    //                                              n = 16*j + l16.
    const float NEG = -__builtin_inff();
    int mrow[8], mcol[8];
    #pragma unroll
    for (int r = 0; r < 8; ++r) mrow[r] = m1row[16 * wave + 8 * half + r];
    #pragma unroll
    for (int j = 0; j < 8; ++j) mcol[j] = m2row[16 * j + l16];

    // ---- masked row max (over valid columns) ----
    float rm[8];
    #pragma unroll
    for (int r = 0; r < 8; ++r) {
        float v = NEG;
        #pragma unroll
        for (int j = 0; j < 8; ++j)
            if (mcol[j]) v = fmaxf(v, acc[j][r]);
        rm[r] = v;
    }
    #pragma unroll
    for (int d = 1; d < 16; d <<= 1) {
        #pragma unroll
        for (int r = 0; r < 8; ++r)
            rm[r] = fmaxf(rm[r], __shfl_xor(rm[r], d, 32));
    }
    if (l16 == 0) {
        float p = 0.f;
        #pragma unroll
        for (int r = 0; r < 8; ++r)
            if (mrow[r]) p += rm[r];
        rowpart[wave * 2 + half] = p;
    }

    // ---- masked column max (over valid rows) ----
    float cm[8];
    #pragma unroll
    for (int j = 0; j < 8; ++j) {
        float v = NEG;
        #pragma unroll
        for (int r = 0; r < 8; ++r)
            if (mrow[r]) v = fmaxf(v, acc[j][r]);
        cm[j] = v;
    }
    #pragma unroll
    for (int j = 0; j < 8; ++j)
        cm[j] = fmaxf(cm[j], __shfl_xor(cm[j], 16, 32));
    if (half == 0) {
        #pragma unroll
        for (int j = 0; j < 8; ++j)
            colpart[wave][16 * j + l16] = cm[j];
    }
    __syncthreads();

    if (tid < SD) {
        float v = colpart[0][tid];
        #pragma unroll
        for (int w = 1; w < 8; ++w) v = fmaxf(v, colpart[w][tid]);
        colval[tid] = m2row[tid] ? v : 0.f;
    }
    __syncthreads();

    if (tid == 0) {
        float s1 = 0.f;
        #pragma unroll
        for (int i = 0; i < 16; ++i) s1 += rowpart[i];
        float s2 = 0.f;
        for (int n = 0; n < SD; ++n) s2 += colval[n];
        out[b1 * BD + b2] = 0.5f * (s1 / (float)s_cnt1 + s2 / (float)s_cnt2);
    }
}

// ---------------------------------------------------------------------------
extern "C" void kernel_launch(void* const* d_in, const int* in_sizes, int n_in,
                              void* d_out, int out_size, void* d_ws, size_t ws_size,
                              hipStream_t stream)
{
    const float* x1    = (const float*)d_in[0];
    const int*   mask1 = (const int*)  d_in[1];
    const float* x2    = (const float*)d_in[2];
    const int*   mask2 = (const int*)  d_in[3];
    float*       out   = (float*)d_out;

    _Float16* y1 = (_Float16*)d_ws;
    _Float16* y2 = y1 + (size_t)BD * SD * HD;

    normalize_f16_kernel<<<BD * SD, 256, 0, stream>>>(x1, y1);
    normalize_f16_kernel<<<BD * SD, 256, 0, stream>>>(x2, y2);

    dim3 grid(BD, BD);
    rwmd_pair_kernel<<<grid, 256, 0, stream>>>(y1, y2, mask1, mask2, out);
}
